// InvariantPointAttention_12867722018817
// MI455X (gfx1250) — compile-verified, matrix operand order checked by reading, not run on hardware
//
#include <hip/hip_runtime.h>
#include <hip/hip_bf16.h>
#include <math.h>

// Problem constants (match reference)
#define H_   12
#define CH   16
#define CZ   128
#define CS   384
#define NB   2
#define NN   512
#define CCAT (H_ * (CH + CZ))   // 1728

typedef __attribute__((ext_vector_type(16))) __bf16 v16bf;
typedef __attribute__((ext_vector_type(8)))  float  v8f;

// K index for element e (0..15) of a 16-bit A/B WMMA operand, given the
// lane-half base (0 or 8).  Lane 0-15: K = {0..7, 16..23}; lane 16-31:
// K = {8..15, 24..31}  (CDNA5 ISA 7.12.2, 16-bit A 16x32 layout).
// Note e=0..7 -> K = kbase+e (contiguous), e=8..15 -> K = kbase+16+(e-8).
__device__ __forceinline__ int kmap(int e, int kbase) {
    return kbase + (e & 7) + ((e >> 3) << 4);
}

// Load the 16 A-operand elements for this lane from a contiguous fp32 row:
// two 8-float contiguous runs -> 4x global_load_b128 + packed bf16 convert.
__device__ __forceinline__ v16bf load_a_rowmajor(const float* __restrict__ row,
                                                 int k0, int kbase) {
    const float4* p0 = reinterpret_cast<const float4*>(row + k0 + kbase);
    const float4* p1 = reinterpret_cast<const float4*>(row + k0 + kbase + 16);
    const float4 a0 = p0[0], a1 = p0[1];
    const float4 a2 = p1[0], a3 = p1[1];
    v16bf a;
    a[0]  = (__bf16)a0.x; a[1]  = (__bf16)a0.y; a[2]  = (__bf16)a0.z; a[3]  = (__bf16)a0.w;
    a[4]  = (__bf16)a1.x; a[5]  = (__bf16)a1.y; a[6]  = (__bf16)a1.z; a[7]  = (__bf16)a1.w;
    a[8]  = (__bf16)a2.x; a[9]  = (__bf16)a2.y; a[10] = (__bf16)a2.z; a[11] = (__bf16)a2.w;
    a[12] = (__bf16)a3.x; a[13] = (__bf16)a3.y; a[14] = (__bf16)a3.z; a[15] = (__bf16)a3.w;
    return a;
}

// ---------------------------------------------------------------------------
// Generic C = A(MxK) @ W(KxN) + bias, fp32 in/out, bf16 WMMA accumulation.
// One wave per 16x16 tile.  K % 32 == 0, M % 16 == 0, N % 16 == 0.
// Used for: QKV projections (K=384) and output projection (K=1728).
// ---------------------------------------------------------------------------
__global__ void gemm_bias_bf16(const float* __restrict__ A,
                               const float* __restrict__ W,
                               const float* __restrict__ bias,
                               float* __restrict__ C,
                               int M, int K, int Nc) {
    const int lane  = threadIdx.x & 31;
    const int l     = lane & 15;
    const int half  = lane >> 4;
    const int kbase = half << 3;
    const int m0    = blockIdx.x << 4;
    const int n0    = blockIdx.y << 4;

    v8f acc = {};
    const float* arow = A + (size_t)(m0 + l) * K;
    for (int k0 = 0; k0 < K; k0 += 32) {
        v16bf a = load_a_rowmajor(arow, k0, kbase);
        v16bf b;
#pragma unroll
        for (int e = 0; e < 16; ++e) {
            const int kk = kmap(e, kbase);
            b[e] = (__bf16)W[(size_t)(k0 + kk) * Nc + (n0 + l)];   // lane-coalesced
        }
        acc = __builtin_amdgcn_wmma_f32_16x16x32_bf16(false, a, false, b,
                                                      (short)0, acc, false, false);
    }
#pragma unroll
    for (int r = 0; r < 8; ++r) {
        const int m = m0 + r + (half << 3);
        C[(size_t)m * Nc + n0 + l] = acc[r] + bias[n0 + l];
    }
}

// ---------------------------------------------------------------------------
// logits[b,i,j,h] = z[b,i,j,:] @ Wb[:,h] + bb[h]
// Treat z as (B*N*N x 128) GEMM against Wb (128 x 12, N padded to 16).
// First (HBM-bound) pass over the 256 MB z tensor: A-side streamed with
// 4x b128 loads per lane per K-step.
// ---------------------------------------------------------------------------
__global__ void zbias_kernel(const float* __restrict__ z,
                             const float* __restrict__ Wb,
                             const float* __restrict__ bb,
                             float* __restrict__ logits) {
    const int lane  = threadIdx.x & 31;
    const int l     = lane & 15;
    const int half  = lane >> 4;
    const int kbase = half << 3;
    const size_t m0 = (size_t)blockIdx.x << 4;

    const float* arow = z + (m0 + l) * CZ;
    __builtin_prefetch(arow + 16 * CZ, 0, 1);   // next row-tile of z

    v8f acc = {};
    for (int k0 = 0; k0 < CZ; k0 += 32) {
        v16bf a = load_a_rowmajor(arow, k0, kbase);
        v16bf b;
#pragma unroll
        for (int e = 0; e < 16; ++e) {
            const int kk = kmap(e, kbase);
            b[e] = (l < H_) ? (__bf16)Wb[(size_t)(k0 + kk) * H_ + l] : (__bf16)0.0f;
        }
        acc = __builtin_amdgcn_wmma_f32_16x16x32_bf16(false, a, false, b,
                                                      (short)0, acc, false, false);
    }
    if (l < H_) {
#pragma unroll
        for (int r = 0; r < 8; ++r) {
            const size_t m = m0 + r + (half << 3);
            logits[m * H_ + l] = acc[r] + bb[l];
        }
    }
}

// ---------------------------------------------------------------------------
// logits += q.k^T / sqrt(CH) + multiscale_bias(d(i,j)); apply mask.
// One wave per (b,h, 16x16 (i,j) tile).  K=16 padded to 32; each lane's
// valid 8 K-values are contiguous -> 2x b128 loads per operand.
// ---------------------------------------------------------------------------
__global__ void qk_bias_kernel(const float* __restrict__ q,
                               const float* __restrict__ k,
                               const float* __restrict__ trans,
                               const unsigned char* __restrict__ mask,
                               float* __restrict__ logits) {
    const int lane  = threadIdx.x & 31;
    const int l     = lane & 15;
    const int half  = lane >> 4;
    const int kbase = half << 3;
    const int i0    = blockIdx.x << 4;
    const int j0    = blockIdx.y << 4;
    const int b     = blockIdx.z / H_;
    const int h     = blockIdx.z % H_;

    const float* qrow = q + (((size_t)b * NN + i0 + l) * H_ + h) * CH + kbase;
    const float* krow = k + (((size_t)b * NN + j0 + l) * H_ + h) * CH + kbase;
    const float4 q0 = ((const float4*)qrow)[0], q1 = ((const float4*)qrow)[1];
    const float4 k0v = ((const float4*)krow)[0], k1v = ((const float4*)krow)[1];

    v16bf a = {}, bm = {};
    a[0] = (__bf16)q0.x; a[1] = (__bf16)q0.y; a[2] = (__bf16)q0.z; a[3] = (__bf16)q0.w;
    a[4] = (__bf16)q1.x; a[5] = (__bf16)q1.y; a[6] = (__bf16)q1.z; a[7] = (__bf16)q1.w;
    bm[0] = (__bf16)k0v.x; bm[1] = (__bf16)k0v.y; bm[2] = (__bf16)k0v.z; bm[3] = (__bf16)k0v.w;
    bm[4] = (__bf16)k1v.x; bm[5] = (__bf16)k1v.y; bm[6] = (__bf16)k1v.z; bm[7] = (__bf16)k1v.w;
    // elements 8..15 (K=16..31) stay zero: K padded from 16 to 32

    v8f acc = {};
    acc = __builtin_amdgcn_wmma_f32_16x16x32_bf16(false, a, false, bm,
                                                  (short)0, acc, false, false);

    const int j = j0 + l;
    const float tx = trans[((size_t)b * NN + j) * 3 + 0];
    const float ty = trans[((size_t)b * NN + j) * 3 + 1];
    const float tz = trans[((size_t)b * NN + j) * 3 + 2];
    const bool  mj = mask[(size_t)b * NN + j] != 0;

#pragma unroll
    for (int r = 0; r < 8; ++r) {
        const int i = i0 + r + (half << 3);
        const float dx = trans[((size_t)b * NN + i) * 3 + 0] - tx;
        const float dy = trans[((size_t)b * NN + i) * 3 + 1] - ty;
        const float dz = trans[((size_t)b * NN + i) * 3 + 2] - tz;
        const float d  = sqrtf(dx * dx + dy * dy + dz * dz);
        const float ms = (d <= 5.0f) ? 1.0f : ((d <= 15.0f) ? 0.3f : 0.05f);
        const bool  mk = mj && (mask[(size_t)b * NN + i] != 0);
        const size_t idx = (((size_t)b * NN + i) * NN + j) * H_ + h;
        logits[idx] = mk ? (logits[idx] + acc[r] * 0.25f + ms) : -1e9f;
    }
}

// ---------------------------------------------------------------------------
// Softmax over j (stride H_) and convert to bf16.  One wave per (b,i,h).
// ---------------------------------------------------------------------------
__global__ void softmax_kernel(const float* __restrict__ logits,
                               __bf16* __restrict__ w) {
    const int lane = threadIdx.x & 31;
    const int row  = blockIdx.x;          // (b*N + i)*H + h
    const int bi   = row / H_;
    const int h    = row % H_;
    const size_t base = (size_t)bi * NN * H_ + h;

    float mx = -3.402823e38f;
    for (int j = lane; j < NN; j += 32)
        mx = fmaxf(mx, logits[base + (size_t)j * H_]);
    for (int off = 16; off; off >>= 1)
        mx = fmaxf(mx, __shfl_xor(mx, off, 32));

    float sum = 0.0f;
    for (int j = lane; j < NN; j += 32)
        sum += __expf(logits[base + (size_t)j * H_] - mx);
    for (int off = 16; off; off >>= 1)
        sum += __shfl_xor(sum, off, 32);

    const float inv = 1.0f / sum;
    for (int j = lane; j < NN; j += 32) {
        const float e = __expf(logits[base + (size_t)j * H_] - mx);
        w[base + (size_t)j * H_] = (__bf16)(e * inv);
    }
}

// ---------------------------------------------------------------------------
// scalar[b,i,h,:] = sum_j w[b,i,j,h] * v[b,j,h,:]  -> concat[..., 0:16]
// One wave per (b,h, 16-row i tile).  M=16 (i), N=16 (C_H), K=512 (j).
// w gather is stride-H_ (cannot vectorize); v loads are lane-coalesced.
// ---------------------------------------------------------------------------
__global__ void scalar_av_kernel(const __bf16* __restrict__ w,
                                 const float* __restrict__ v,
                                 float* __restrict__ concat) {
    const int lane  = threadIdx.x & 31;
    const int l     = lane & 15;
    const int half  = lane >> 4;
    const int kbase = half << 3;
    const int i0    = blockIdx.x << 4;
    const int b     = blockIdx.y / H_;
    const int h     = blockIdx.y % H_;

    v8f acc = {};
    for (int k0 = 0; k0 < NN; k0 += 32) {
        v16bf a, bm;
#pragma unroll
        for (int e = 0; e < 16; ++e) {
            const int kk = k0 + kmap(e, kbase);
            a[e]  = w[(((size_t)b * NN + i0 + l) * NN + kk) * H_ + h];
            bm[e] = (__bf16)v[(((size_t)b * NN + kk) * H_ + h) * CH + l];
        }
        acc = __builtin_amdgcn_wmma_f32_16x16x32_bf16(false, a, false, bm,
                                                      (short)0, acc, false, false);
    }
#pragma unroll
    for (int r = 0; r < 8; ++r) {
        const int i = i0 + r + (half << 3);
        concat[((size_t)b * NN + i) * CCAT + h * (CH + CZ) + l] = acc[r];
    }
}

// ---------------------------------------------------------------------------
// pair[b,i,h,:] = sum_j w[b,i,j,h] * z[b,i,j,:]  -> concat[..., 16:144]
// One wave per (b,i, 16-col tile of C_Z).  M=16 (h, 12 valid), K=512, N=16.
// Second (HBM-bound) pass over z; z loads are lane-coalesced (consecutive
// lanes -> consecutive dwords), prefetch hints the next K-tile.
// ---------------------------------------------------------------------------
__global__ void pair_wz_kernel(const __bf16* __restrict__ w,
                               const float* __restrict__ z,
                               float* __restrict__ concat) {
    const int lane  = threadIdx.x & 31;
    const int l     = lane & 15;
    const int half  = lane >> 4;
    const int kbase = half << 3;
    const int c0    = blockIdx.x << 4;
    const int bi    = blockIdx.y;        // b*N + i
    const size_t rowbase = (size_t)bi * NN;

    v8f acc = {};
    for (int k0 = 0; k0 < NN; k0 += 32) {
        __builtin_prefetch(&z[(rowbase + k0 + 32) * CZ + c0 + l], 0, 1);
        v16bf a, bm;
#pragma unroll
        for (int e = 0; e < 16; ++e) {
            const int kk = k0 + kmap(e, kbase);
            a[e]  = (l < H_) ? w[(rowbase + kk) * H_ + l] : (__bf16)0.0f;
            bm[e] = (__bf16)z[(rowbase + kk) * CZ + c0 + l];
        }
        acc = __builtin_amdgcn_wmma_f32_16x16x32_bf16(false, a, false, bm,
                                                      (short)0, acc, false, false);
    }
#pragma unroll
    for (int r = 0; r < 8; ++r) {
        const int hh = r + (half << 3);
        if (hh < H_)
            concat[(size_t)bi * CCAT + hh * (CH + CZ) + CH + c0 + l] = acc[r];
    }
}

// ---------------------------------------------------------------------------
extern "C" void kernel_launch(void* const* d_in, const int* in_sizes, int n_in,
                              void* d_out, int out_size, void* d_ws, size_t ws_size,
                              hipStream_t stream) {
    const float* s    = (const float*)d_in[0];
    const float* z    = (const float*)d_in[1];
    const float* tr   = (const float*)d_in[2];
    // d_in[3] = rotations : unused by the reference computation
    const unsigned char* mask = (const unsigned char*)d_in[4];
    const float* Wq   = (const float*)d_in[5];
    const float* bq   = (const float*)d_in[6];
    const float* Wk   = (const float*)d_in[7];
    const float* bk   = (const float*)d_in[8];
    const float* Wv   = (const float*)d_in[9];
    const float* bv   = (const float*)d_in[10];
    const float* Wb   = (const float*)d_in[11];
    const float* bb   = (const float*)d_in[12];
    const float* Wout = (const float*)d_in[13];
    const float* bout = (const float*)d_in[14];
    float* out = (float*)d_out;

    // Workspace carve-up (all regions fully overwritten every call).
    const size_t qkv_elems = (size_t)NB * NN * H_ * CH;      // 196608 f32
    const size_t lg_elems  = (size_t)NB * NN * NN * H_;      // 6291456
    float*  q      = (float*)d_ws;
    float*  k      = q + qkv_elems;
    float*  v      = k + qkv_elems;
    float*  logits = v + qkv_elems;                          // 24 MB f32
    __bf16* wsm    = (__bf16*)(logits + lg_elems);           // 12 MB bf16
    float*  concat = (float*)(wsm + lg_elems);               // 6.75 MB f32

    const dim3 blk(32);  // one wave32 per block; EXEC all-ones for WMMA

    // 1) QKV projections: (1024 x 384) @ (384 x 192) x3
    gemm_bias_bf16<<<dim3((NB * NN) / 16, (H_ * CH) / 16), blk, 0, stream>>>(
        s, Wq, bq, q, NB * NN, CS, H_ * CH);
    gemm_bias_bf16<<<dim3((NB * NN) / 16, (H_ * CH) / 16), blk, 0, stream>>>(
        s, Wk, bk, k, NB * NN, CS, H_ * CH);
    gemm_bias_bf16<<<dim3((NB * NN) / 16, (H_ * CH) / 16), blk, 0, stream>>>(
        s, Wv, bv, v, NB * NN, CS, H_ * CH);

    // 2) Pair bias: stream z once, (524288 x 128) @ (128 x 12)
    zbias_kernel<<<dim3((NB * NN * NN) / 16), blk, 0, stream>>>(z, Wb, bb, logits);

    // 3) QK^T / 4 + multiscale bias + mask (fused epilogue)
    qk_bias_kernel<<<dim3(NN / 16, NN / 16, NB * H_), blk, 0, stream>>>(
        q, k, tr, mask, logits);

    // 4) Softmax over j -> bf16 weights
    softmax_kernel<<<dim3(NB * NN * H_), blk, 0, stream>>>(logits, wsm);

    // 5) scalar = w @ v
    scalar_av_kernel<<<dim3(NN / 16, NB * H_), blk, 0, stream>>>(wsm, v, concat);

    // 6) pair = w @ z (second stream of z)
    pair_wz_kernel<<<dim3(CZ / 16, NB * NN), blk, 0, stream>>>(wsm, z, concat);

    // 7) out = concat (1024 x 1728) @ Wout (1728 x 384) + bout
    gemm_bias_bf16<<<dim3((NB * NN) / 16, CS / 16), blk, 0, stream>>>(
        concat, Wout, bout, out, NB * NN, CCAT, CS);
}
